// MultiHeadAttention_28724741276186
// MI455X (gfx1250) — matrix operand order repaired and checked by reading, not compiled, not run on hardware
//
#include <hip/hip_runtime.h>
#include <hip/hip_bf16.h>

#define D_MODEL  1024
#define N_HEADS  16
#define HEAD_DIM 64
#define BATCH    2
#define SEQ      2048
#define NTOK     (BATCH * SEQ)

typedef __attribute__((ext_vector_type(16))) __bf16 v16bf;
typedef __attribute__((ext_vector_type(8)))  __bf16 v8bf;
typedef __attribute__((ext_vector_type(2)))  __bf16 v2bf;
typedef __attribute__((ext_vector_type(8)))  float  v8f;

__device__ __forceinline__ __bf16 f2bf(float f) { return (__bf16)f; }  // native cvt

// Swap bits 3<->4 of k (0..31): A-fragment swizzle so each lane-half's 16
// elements are contiguous (pos 0-15 -> K{0-7,16-23}, pos 16-31 -> K{8-15,24-31}).
__device__ __forceinline__ int swz32(int k) {
  return (k & 7) | ((k & 8) << 1) | ((k & 16) >> 1);
}

// Async global->LDS copy of 32 bytes (two B128 ops), tracked by ASYNCcnt.
// Same inst_offset applies to both LDS and global sides (ISA 15.18.3 op 98).
__device__ __forceinline__ void async_copy32(const __bf16* lds_dst,
                                             const __bf16* gsrc) {
  unsigned dst = (unsigned)(size_t)lds_dst;            // low 32b = LDS offset
  unsigned long long src = (unsigned long long)(size_t)gsrc;
  asm volatile("global_load_async_to_lds_b128 %0, %1, off\n\t"
               "global_load_async_to_lds_b128 %0, %1, off offset:16"
               :: "v"(dst), "v"(src) : "memory");
}
__device__ __forceinline__ void wait_async0() {
  asm volatile("s_wait_asynccnt 0x0" ::: "memory");
}

// ---------------------------------------------------------------------------
// Tiled GEMM: C[M,N] (f32) = A[M,K] @ W[K,N], A is f32 or bf16, W is f32.
// Block = 256 thr (8 waves), tile 128x128, BK=32. Wave tile 64x32 = 4x2 WMMA.
// Software-pipelined: next tile's global loads are in flight during WMMA.
// As is stored K-swizzled so A-frags are single contiguous 32B LDS loads.
// W^T staging packs K-pairs -> v_cvt_pk_bf16_f32 + ds_store_b32.
// ---------------------------------------------------------------------------
template <typename AT>
__global__ __launch_bounds__(256) void gemm_wmma_kernel(
    const AT* __restrict__ A, const float* __restrict__ W,
    float* __restrict__ C, int M, int N, int K)
{
  __shared__ __attribute__((aligned(32))) __bf16 As[128 * 32];   // [m][swz(k)]
  __shared__ __attribute__((aligned(32))) __bf16 Bs[128 * 32];   // [n][swz(k)] (W^T tile)

  const int tid  = threadIdx.x;
  const int lane = tid & 31;
  const int wv   = tid >> 5;        // 0..7
  const int wm   = wv >> 2;         // 0..1
  const int wn   = wv & 3;          // 0..3
  const int half = lane >> 4;       // 0..1
  const int l16  = lane & 15;
  const int gm   = blockIdx.y * 128;
  const int gn   = blockIdx.x * 128;

  v8f acc[4][2] = {};

  const int arow  = tid >> 1;          // 0..127
  const int aks   = (tid & 1) * 16;    // K chunk: 0 or 16
  const int abase = (tid & 1) * 8;     // swizzled dst base: 0 or 8
  const int wkp   = (tid & 15) * 2;    // even K pair: 0..30
  const int wns   = (tid >> 4) * 8;    // 8-wide N segment: 0..120

  const AT*    aptr  = A + (size_t)(gm + arow) * K + aks;
  const float* wptr0 = W + (size_t)wkp * N + gn + wns;        // row k
  const float* wptr1 = wptr0 + N;                             // row k+1

  float4 areg[4];      // staged A (f32 path)
  v8bf   aregh[2];     // staged A (bf16 path)
  float4 wreg[4];      // staged W: row0 [0-3],[4-7], row1 [0-3],[4-7]

  auto load_stage = [&](int k0) {
    if constexpr (sizeof(AT) == 4) {
      const float4* s4 = (const float4*)(aptr + k0);
      areg[0] = s4[0]; areg[1] = s4[1]; areg[2] = s4[2]; areg[3] = s4[3];
    } else {
      const v8bf* s8 = (const v8bf*)(aptr + k0);
      aregh[0] = s8[0]; aregh[1] = s8[1];
    }
    const float4* w0 = (const float4*)(wptr0 + (size_t)k0 * N);
    const float4* w1 = (const float4*)(wptr1 + (size_t)k0 * N);
    wreg[0] = w0[0]; wreg[1] = w0[1]; wreg[2] = w1[0]; wreg[3] = w1[1];
  };

  auto store_stage = [&]() {
    __bf16* dst = &As[arow * 32 + abase];  // [abase..+8) and [abase+16..+24)
    if constexpr (sizeof(AT) == 4) {
      dst[0]=f2bf(areg[0].x); dst[1]=f2bf(areg[0].y); dst[2]=f2bf(areg[0].z); dst[3]=f2bf(areg[0].w);
      dst[4]=f2bf(areg[1].x); dst[5]=f2bf(areg[1].y); dst[6]=f2bf(areg[1].z); dst[7]=f2bf(areg[1].w);
      dst[16]=f2bf(areg[2].x); dst[17]=f2bf(areg[2].y); dst[18]=f2bf(areg[2].z); dst[19]=f2bf(areg[2].w);
      dst[20]=f2bf(areg[3].x); dst[21]=f2bf(areg[3].y); dst[22]=f2bf(areg[3].z); dst[23]=f2bf(areg[3].w);
    } else {
      *(v8bf*)dst        = aregh[0];
      *(v8bf*)(dst + 16) = aregh[1];
    }
    float r0[8] = {wreg[0].x,wreg[0].y,wreg[0].z,wreg[0].w,
                   wreg[1].x,wreg[1].y,wreg[1].z,wreg[1].w};
    float r1[8] = {wreg[2].x,wreg[2].y,wreg[2].z,wreg[2].w,
                   wreg[3].x,wreg[3].y,wreg[3].z,wreg[3].w};
    #pragma unroll
    for (int j = 0; j < 8; ++j) {
      v2bf p; p[0] = f2bf(r0[j]); p[1] = f2bf(r1[j]);   // (k, k+1) pair
      *(v2bf*)&Bs[(wns + j) * 32 + swz32(wkp)] = p;     // packed b32 store
    }
  };

  load_stage(0);
  for (int k0 = 0; k0 < K; k0 += 32) {
    __syncthreads();                 // previous tile's readers done (no-op 1st iter)
    store_stage();
    __syncthreads();
    if (k0 + 32 < K) load_stage(k0 + 32);   // overlap with WMMA below

    v16bf af[4];
    #pragma unroll
    for (int mi = 0; mi < 4; ++mi)
      af[mi] = *(const v16bf*)&As[(wm * 64 + mi * 16 + l16) * 32 + half * 16];
    v16bf bfr[2];
    #pragma unroll
    for (int ni = 0; ni < 2; ++ni)
      bfr[ni] = *(const v16bf*)&Bs[(wn * 32 + ni * 16 + l16) * 32 + half * 16];

    #pragma unroll
    for (int mi = 0; mi < 4; ++mi)
      #pragma unroll
      for (int ni = 0; ni < 2; ++ni)
        acc[mi][ni] = __builtin_amdgcn_wmma_f32_16x16x32_bf16(
            false, af[mi], false, bfr[ni], (short)0, acc[mi][ni], false, false);
  }

  // ---- store: D element (M,N): M = r + 8*half, N = l16
  #pragma unroll
  for (int mi = 0; mi < 4; ++mi)
    #pragma unroll
    for (int ni = 0; ni < 2; ++ni)
      #pragma unroll
      for (int r = 0; r < 8; ++r)
        C[(size_t)(gm + wm * 64 + mi * 16 + r + half * 8) * N +
          (gn + wn * 32 + ni * 16 + l16)] = acc[mi][ni][r];
}

// ---------------------------------------------------------------------------
// RMSNorm + RoPE + repack. One wave per (b,h,t) row; lane i owns (i, i+32).
// Qh: bf16 [B,H,T,64], K-swizzled per 32-chunk (A-frag consumer).
// Kh: bf16 [B,H,T,64] K-swizzled per 32-chunk (B-frag consumer).
// Vt: bf16 [B,H,64,T] transposed (B-frag consumer, contiguous keys).
// ---------------------------------------------------------------------------
__global__ __launch_bounds__(256) void norm_rope_pack_kernel(
    const float* __restrict__ Q0, const float* __restrict__ K0,
    const float* __restrict__ V0, const float* __restrict__ qw,
    const float* __restrict__ kw, __bf16* __restrict__ Qh,
    __bf16* __restrict__ Kh, __bf16* __restrict__ Vt)
{
  const int lane = threadIdx.x & 31;
  const int wv   = threadIdx.x >> 5;
  const int rid  = blockIdx.x * 8 + wv;        // 0 .. B*H*T-1
  const int b = rid / (N_HEADS * SEQ);
  const int rem = rid % (N_HEADS * SEQ);
  const int h = rem / SEQ;
  const int t = rem % SEQ;

  const size_t src = (size_t)(b * SEQ + t) * D_MODEL + h * HEAD_DIM;
  const int i = lane;
  float q1 = Q0[src + i], q2 = Q0[src + i + 32];
  float k1 = K0[src + i], k2 = K0[src + i + 32];
  float v1 = V0[src + i], v2 = V0[src + i + 32];

  float sq = q1 * q1 + q2 * q2;
  float sk = k1 * k1 + k2 * k2;
  #pragma unroll
  for (int off = 1; off < 32; off <<= 1) {
    sq += __shfl_xor(sq, off, 32);
    sk += __shfl_xor(sk, off, 32);
  }
  const float rq = rsqrtf(sq * (1.0f / 64.0f) + 1e-6f);
  const float rk = rsqrtf(sk * (1.0f / 64.0f) + 1e-6f);
  q1 = q1 * rq * qw[i]; q2 = q2 * rq * qw[i + 32];
  k1 = k1 * rk * kw[i]; k2 = k2 * rk * kw[i + 32];

  // inv_freq[i] = 10000^(-2i/64)
  const float invf = __expf(-9.2103403719761836f * ((float)(2 * i) * (1.0f / 64.0f)));
  const float ang = (float)t * invf;
  const float c = cosf(ang), s = sinf(ang);
  const float qo1 = q1 * c - q2 * s, qo2 = q2 * c + q1 * s;
  const float ko1 = k1 * c - k2 * s, ko2 = k2 * c + k1 * s;

  const size_t dst = (size_t)((b * N_HEADS + h) * SEQ + t) * HEAD_DIM;
  const int pos = swz32(i);
  Qh[dst + pos] = f2bf(qo1); Qh[dst + 32 + pos] = f2bf(qo2);   // swizzled
  Kh[dst + pos] = f2bf(ko1); Kh[dst + 32 + pos] = f2bf(ko2);   // swizzled
  const size_t vdst = (size_t)(b * N_HEADS + h) * HEAD_DIM * SEQ;
  Vt[vdst + (size_t)i * SEQ + t]        = f2bf(v1);
  Vt[vdst + (size_t)(i + 32) * SEQ + t] = f2bf(v2);
}

// ---------------------------------------------------------------------------
// Flash attention, unnormalized softmax (RMSNorm bounds |logit| <= 8 so exp
// cannot overflow; running max unnecessary). Row sums via P @ ones WMMA.
// Block = 128 thr (4 waves), 64 queries/block, 16/wave; 32-key blocks.
// K/V tiles double-buffered in LDS, filled by GLOBAL_LOAD_ASYNC_TO_LDS_B128
// (ASYNCcnt) issued one block ahead, overlapping the DMA with compute.
// ---------------------------------------------------------------------------
__global__ __launch_bounds__(128) void attn_wmma_kernel(
    const __bf16* __restrict__ Qh, const __bf16* __restrict__ Kh,
    const __bf16* __restrict__ Vt, __bf16* __restrict__ Ao)
{
  __shared__ __attribute__((aligned(32))) __bf16 Klds[2][32 * 64];  // [buf][key][hd]
  __shared__ __attribute__((aligned(32))) __bf16 Vlds[2][64 * 32];  // [buf][hd][key]
  __shared__ __attribute__((aligned(32))) __bf16 Plds[4 * 512];     // per-wave 16x32, plain K order

  const int tid  = threadIdx.x;
  const int lane = tid & 31;
  const int wv   = tid >> 5;
  const int half = lane >> 4;
  const int l16  = lane & 15;
  const int b = blockIdx.z, h = blockIdx.y;
  const int qbase = blockIdx.x * 64 + wv * 16;
  const float scale = 0.125f;   // 1/sqrt(64)

  const __bf16* Qp = Qh + (size_t)(b * N_HEADS + h) * SEQ * HEAD_DIM;
  const __bf16* Kp = Kh + (size_t)(b * N_HEADS + h) * SEQ * HEAD_DIM;
  const __bf16* Vp = Vt + (size_t)(b * N_HEADS + h) * HEAD_DIM * SEQ;

  // Q A-frags: Qh is pre-swizzled -> one contiguous 32B load per chunk
  v16bf qa[2];
  #pragma unroll
  for (int c = 0; c < 2; ++c)
    qa[c] = *(const v16bf*)(Qp + (size_t)(qbase + l16) * HEAD_DIM + c * 32 + half * 16);

  // all-ones B fragment for matrix-engine row sums
  v16bf onesf;
  #pragma unroll
  for (int e = 0; e < 16; ++e)
    onesf[e] = __builtin_bit_cast(__bf16, (unsigned short)0x3F80);

  v8f o[4] = {};
  v8f osum = {};

  // cooperative K/V staging assignments (128 threads x 32B segments)
  const int krow = tid >> 2, kseg = (tid & 3) * 16;   // 32 keys x 4 segs
  const int vrow = tid >> 1, vseg = (tid & 1) * 16;   // 64 hd   x 2 segs
  const __bf16* kap = Kp + (size_t)krow * HEAD_DIM + kseg;
  const __bf16* vap = Vp + (size_t)vrow * SEQ + vseg;

  auto stage_async = [&](int kbase, int buf) {
    async_copy32(&Klds[buf][krow * 64 + kseg], kap + (size_t)kbase * HEAD_DIM);
    async_copy32(&Vlds[buf][vrow * 32 + vseg], vap + kbase);
  };

  const int nkb = blockIdx.x * 2 + 2;   // uniform across block
  stage_async(0, 0);
  for (int kb = 0; kb < nkb; ++kb) {
    const int kbase = kb * 32;
    const __bf16* Kb = &Klds[kb & 1][0];
    const __bf16* Vb = &Vlds[kb & 1][0];

    wait_async0();                         // our async writes for this buf done
    __syncthreads();                       // everyone's done (and prior readers)
    if (kb + 1 < nkb) stage_async(kbase + 32, (kb + 1) & 1);  // DMA next buf

    // ---- S = Q K^T for 2 key sub-tiles of 16 (K frags from LDS)
    v8f s[2];
    #pragma unroll
    for (int ti = 0; ti < 2; ++ti) {
      v8f z = {};
      #pragma unroll
      for (int c = 0; c < 2; ++c) {
        v16bf kf = *(const v16bf*)&Kb[(ti * 16 + l16) * 64 + c * 32 + half * 16];
        z = __builtin_amdgcn_wmma_f32_16x16x32_bf16(false, qa[c], false, kf,
                                                    (short)0, z, false, false);
      }
      s[ti] = z;
    }

    // ---- p = exp(s*scale), causal mask only on diagonal/upper blocks
    float p0[8], p1[8];
    if (kbase + 32 > qbase) {            // wave-uniform branch (no barriers inside)
      #pragma unroll
      for (int r = 0; r < 8; ++r) {
        const int q = qbase + r + half * 8;
        const float s0 = (kbase + l16 > q)      ? -3.0e38f : s[0][r] * scale;
        const float s1 = (kbase + 16 + l16 > q) ? -3.0e38f : s[1][r] * scale;
        p0[r] = __expf(s0);
        p1[r] = __expf(s1);
      }
    } else {
      #pragma unroll
      for (int r = 0; r < 8; ++r) {
        p0[r] = __expf(s[0][r] * scale);
        p1[r] = __expf(s[1][r] * scale);
      }
    }

    // ---- write P (bf16, plain K order) to wave-private LDS
    #pragma unroll
    for (int r = 0; r < 8; ++r) {
      const int m = r + half * 8;
      Plds[wv * 512 + m * 32 + l16]      = f2bf(p0[r]);
      Plds[wv * 512 + m * 32 + 16 + l16] = f2bf(p1[r]);
    }

    // ---- P as A-frag: one contiguous 32B LDS load (in-order DS per wave)
    v16bf pa = *(const v16bf*)&Plds[wv * 512 + l16 * 32 + half * 16];

    // ---- O += P @ V ; rowsum += P @ ones (V frags from LDS)
    #pragma unroll
    for (int t4 = 0; t4 < 4; ++t4) {
      v16bf vf = *(const v16bf*)&Vb[(t4 * 16 + l16) * 32 + half * 16];
      o[t4] = __builtin_amdgcn_wmma_f32_16x16x32_bf16(false, pa, false, vf,
                                                      (short)0, o[t4], false, false);
    }
    osum = __builtin_amdgcn_wmma_f32_16x16x32_bf16(false, pa, false, onesf,
                                                   (short)0, osum, false, false);
  }

  // ---- epilogue: divide by row sum, store bf16 [B,T,H*64]
  #pragma unroll
  for (int r = 0; r < 8; ++r) {
    const float inv = 1.0f / osum[r];
    const int q = qbase + r + half * 8;
    #pragma unroll
    for (int t4 = 0; t4 < 4; ++t4) {
      const int hd = t4 * 16 + l16;
      Ao[(size_t)(b * SEQ + q) * D_MODEL + h * HEAD_DIM + hd] = f2bf(o[t4][r] * inv);
    }
  }
}

// ---------------------------------------------------------------------------
extern "C" void kernel_launch(void* const* d_in, const int* in_sizes, int n_in,
                              void* d_out, int out_size, void* d_ws, size_t ws_size,
                              hipStream_t stream) {
  (void)in_sizes; (void)n_in; (void)out_size; (void)ws_size;
  const float* x  = (const float*)d_in[0];
  const float* Wq = (const float*)d_in[1];
  const float* Wk = (const float*)d_in[2];
  const float* Wv = (const float*)d_in[3];
  const float* Wo = (const float*)d_in[4];
  const float* qw = (const float*)d_in[5];
  const float* kw = (const float*)d_in[6];
  float* out = (float*)d_out;

  char* ws = (char*)d_ws;
  const size_t f32mat  = (size_t)NTOK * D_MODEL * sizeof(float);  // 16 MB
  const size_t bf16mat = (size_t)NTOK * D_MODEL * sizeof(short);  // 8 MB
  float*  Q0 = (float*)ws;  ws += f32mat;
  float*  K0 = (float*)ws;  ws += f32mat;
  float*  V0 = (float*)ws;  ws += f32mat;
  __bf16* Qh = (__bf16*)ws; ws += bf16mat;
  __bf16* Kh = (__bf16*)ws; ws += bf16mat;
  __bf16* Vt = (__bf16*)ws; ws += bf16mat;
  __bf16* Ao = (__bf16*)ws; ws += bf16mat;

  dim3 gg(D_MODEL / 128, NTOK / 128);   // (8, 32)
  gemm_wmma_kernel<float><<<gg, 256, 0, stream>>>(x, Wq, Q0, NTOK, D_MODEL, D_MODEL);
  gemm_wmma_kernel<float><<<gg, 256, 0, stream>>>(x, Wk, K0, NTOK, D_MODEL, D_MODEL);
  gemm_wmma_kernel<float><<<gg, 256, 0, stream>>>(x, Wv, V0, NTOK, D_MODEL, D_MODEL);

  norm_rope_pack_kernel<<<BATCH * N_HEADS * SEQ / 8, 256, 0, stream>>>(
      Q0, K0, V0, qw, kw, Qh, Kh, Vt);

  attn_wmma_kernel<<<dim3(SEQ / 64, N_HEADS, BATCH), 128, 0, stream>>>(Qh, Kh, Vt, Ao);

  gemm_wmma_kernel<__bf16><<<gg, 256, 0, stream>>>(Ao, Wo, out, NTOK, D_MODEL, D_MODEL);
}
